// ResidualAttention_61727269978676
// MI455X (gfx1250) — matrix operand order repaired and checked by reading, not compiled, not run on hardware
//
#include <hip/hip_runtime.h>

// Residual attention forward for MI455X (gfx1250), wave32 + WMMA f16.
// B=8, L=S=1024, H=8, E=D=64.

#define B_ 8
#define L_ 1024
#define S_ 1024
#define H_ 8
#define E_ 64
#define D_ 64
#define LT_ (L_ / 16)   // 64 L-tiles
#define ST_ (S_ / 16)   // 64 S-tiles
#define WAVES 4

typedef __attribute__((ext_vector_type(16))) _Float16 v16h;
typedef __attribute__((ext_vector_type(8)))  float    v8f;
typedef __attribute__((ext_vector_type(4)))  float    v4f;

__global__ __launch_bounds__(WAVES * 32)
void resattn_fwd_kernel(const float* __restrict__ q,
                        const float* __restrict__ k,
                        const float* __restrict__ v,
                        const float* __restrict__ Ab,
                        float* __restrict__ outV,
                        float* __restrict__ outA)
{
    // per-wave P staging tile for C-layout -> A-layout transpose (f16), padded
    __shared__ _Float16 pbuf[WAVES][16][34];

    const int tid  = threadIdx.x;
    const int wave = tid >> 5;
    const int lane = tid & 31;
    const int half = lane >> 4;   // 0 or 1
    const int ln   = lane & 15;

    const int t  = blockIdx.x * WAVES + wave;   // tile id, 0..4095
    const int lt = t >> 6;                      // l-tile index (same for whole block)
    const int bh = t & 63;
    const int b  = bh >> 3;
    const int h  = bh & 7;
    const int l0 = lt << 4;

    // ---- Q tile (16 x 64 f32 -> f16) in WMMA A-operand layout, kept in regs
    v16h qa0, qa1;
    {
        const float* qrow = q + (((long)b * L_ + (l0 + ln)) * H_ + h) * E_;
        const int e0 = 8 * half;
#pragma unroll
        for (int i = 0; i < 8; ++i) {
            qa0[i]     = (_Float16)qrow[e0 + i];
            qa0[8 + i] = (_Float16)qrow[e0 + 16 + i];
            qa1[i]     = (_Float16)qrow[e0 + 32 + i];
            qa1[8 + i] = (_Float16)qrow[e0 + 48 + i];
        }
    }

    // base-2 softmax domain: exp(x) == exp2(x * log2e); fold log2e into scores
    const float LOG2E   = 1.44269504088896340736f;
    const float SC_L2E  = 0.125f * LOG2E;   // (1/sqrt(E)) * log2e
    const float NEGBIG  = -1e30f;           // finite stand-in for -inf (NaN-safe)

    // Precompute per-row constants: row index and flat A-row offset (+ln col)
    const long abase_bh = (long)(b * H_ + h) * L_;
    int  lr[8];
    long aoff[8];
#pragma unroll
    for (int r = 0; r < 8; ++r) {
        lr[r]   = l0 + r + 8 * half;
        aoff[r] = (abase_bh + lr[r]) * (long)S_ + ln;
    }

    // score tile for S-tile j: C-layout regs (lane owns rows lr[r], col ln)
    auto score_tile = [&](int s0) -> v8f {
        v16h kb0, kb1;
        const float* krow =
            k + (((long)b * S_ + (s0 + ln)) * H_ + h) * E_ + 8 * half;
#pragma unroll
        for (int i = 0; i < 16; ++i) {
            kb0[i] = (_Float16)krow[i + (i >= 8 ? 8 : 0)];
            kb1[i] = (_Float16)krow[32 + i + (i >= 8 ? 8 : 0)];
        }
        v8f c = {};
        c = __builtin_amdgcn_wmma_f32_16x16x32_f16(false, qa0, false, kb0,
                                                   (short)0, c, false, false);
        c = __builtin_amdgcn_wmma_f32_16x16x32_f16(false, qa1, false, kb1,
                                                   (short)0, c, false, false);
        return c;
    };

    // ===================== pass 1: online softmax stats (base-2) ============
    float mrow[8], lsum[8];
#pragma unroll
    for (int r = 0; r < 8; ++r) { mrow[r] = NEGBIG; lsum[r] = 0.0f; }

    for (int j = 0; j <= lt; ++j) {
        const int s0 = j << 4;
        // Unconditional bias loads issued FIRST: overlap with K loads + WMMA.
        float abuf[8];
#pragma unroll
        for (int r = 0; r < 8; ++r) abuf[r] = Ab[aoff[r] + s0] * LOG2E;
        const v8f c    = score_tile(s0);
        const int scol = s0 + ln;
#pragma unroll
        for (int r = 0; r < 8; ++r) {
            // branchless causal mask; score already in log2 domain
            const float sval =
                (scol > lr[r]) ? NEGBIG : fmaf(c[r], SC_L2E, abuf[r]);
            const float mn = fmaxf(mrow[r], sval);
            lsum[r] = lsum[r] * __builtin_amdgcn_exp2f(mrow[r] - mn) +
                      __builtin_amdgcn_exp2f(sval - mn);
            mrow[r] = mn;
        }
    }

    // reduce stats across the 16-lane half-groups
    float linv[8];
#pragma unroll
    for (int r = 0; r < 8; ++r) {
        float M = mrow[r];
#pragma unroll
        for (int msk = 1; msk < 16; msk <<= 1)
            M = fmaxf(M, __shfl_xor(M, msk, 32));
        float ts = lsum[r] * __builtin_amdgcn_exp2f(mrow[r] - M);
#pragma unroll
        for (int msk = 1; msk < 16; msk <<= 1)
            ts += __shfl_xor(ts, msk, 32);
        mrow[r] = M;
        linv[r] = 1.0f / ts;   // ts >= 1 (max element contributes exp2(0))
    }

    // ===================== pass 2: probs out + P*V =====================
    v8f oacc[4] = {{}, {}, {}, {}};

    // PV over a 32-wide S block starting at sbase; P is in pbuf cols 0..31
    auto do_pv = [&](int sbase) {
        __builtin_amdgcn_wave_barrier();   // keep LDS store->load order
        v16h pa;
        const int e0 = 8 * half;
#pragma unroll
        for (int i = 0; i < 8; ++i) {
            pa[i]     = pbuf[wave][ln][e0 + i];
            pa[8 + i] = pbuf[wave][ln][16 + e0 + i];
        }
        const float* vb_base =
            v + (((long)b * S_ + sbase + 8 * half) * H_ + h) * D_ + ln;
#pragma unroll
        for (int cx = 0; cx < 4; ++cx) {
            v16h vb;
#pragma unroll
            for (int i = 0; i < 16; ++i)
                vb[i] = (_Float16)vb_base[(long)(i + (i >= 8 ? 8 : 0)) * (H_ * D_) + 16 * cx];
            oacc[cx] = __builtin_amdgcn_wmma_f32_16x16x32_f16(
                false, pa, false, vb, (short)0, oacc[cx], false, false);
        }
    };

    for (int j = 0; j <= lt; ++j) {
        const int s0   = j << 4;
        const int slot = j & 1;
        float abuf[8];
#pragma unroll
        for (int r = 0; r < 8; ++r) abuf[r] = Ab[aoff[r] + s0] * LOG2E;
        const v8f c    = score_tile(s0);
        const int scol = s0 + ln;
#pragma unroll
        for (int r = 0; r < 8; ++r) {
            const float sval =
                (scol > lr[r]) ? NEGBIG : fmaf(c[r], SC_L2E, abuf[r]);
            const float p =
                __builtin_amdgcn_exp2f(sval - mrow[r]) * linv[r];
            // streaming write-once tensor: non-temporal to protect L2 reuse
            __builtin_nontemporal_store(p, &outA[aoff[r] + s0]);
            pbuf[wave][r + 8 * half][slot * 16 + ln] = (_Float16)p;
        }
        if (slot == 1) do_pv(s0 - 16);
    }
    if ((lt & 1) == 0) {   // odd tile count: zero-pad slot 1, flush last pair
#pragma unroll
        for (int r = 0; r < 8; ++r)
            pbuf[wave][r + 8 * half][16 + ln] = (_Float16)0.0f;
        do_pv(lt << 4);
    }

    // zero-fill A_new above the diagonal band (masked region, still output).
    // Repacked: lane -> (row = lane>>2, 4-col block = lane&3): two b128
    // NT stores per 16x16 tile (64B-contiguous row groups).
    {
        const int zrow0 = lane >> 2;          // 0..7
        const int zcol  = (lane & 3) << 2;    // 0,4,8,12
        const v4f z4 = {0.0f, 0.0f, 0.0f, 0.0f};
        for (int j = lt + 1; j < ST_; ++j) {
            const int s0 = j << 4;
            float* p0 = &outA[(abase_bh + l0 + zrow0) * (long)S_ + s0 + zcol];
            float* p1 = &outA[(abase_bh + l0 + zrow0 + 8) * (long)S_ + s0 + zcol];
            __builtin_nontemporal_store(z4, (v4f*)p0);
            __builtin_nontemporal_store(z4, (v4f*)p1);
        }
    }

    // write V out: (B, L, H, D), lanes 0..15 cover consecutive d (64B)
#pragma unroll
    for (int cx = 0; cx < 4; ++cx) {
#pragma unroll
        for (int r = 0; r < 8; ++r) {
            outV[(((long)b * L_ + lr[r]) * H_ + h) * D_ + 16 * cx + ln] =
                oacc[cx][r];
        }
    }
}

extern "C" void kernel_launch(void* const* d_in, const int* in_sizes, int n_in,
                              void* d_out, int out_size, void* d_ws, size_t ws_size,
                              hipStream_t stream) {
    (void)in_sizes; (void)n_in; (void)d_ws; (void)ws_size; (void)out_size;
    const float* q  = (const float*)d_in[0];   // (B,L,H,E)
    const float* k  = (const float*)d_in[1];   // (B,S,H,E)
    const float* v  = (const float*)d_in[2];   // (B,S,H,D)
    // d_in[3] = attn_mask (strict-upper causal) -> computed analytically
    const float* Ab = (const float*)d_in[4];   // (B,H,L,S)

    float* outV = (float*)d_out;                          // (B,L,H,D)
    float* outA = outV + (size_t)B_ * L_ * H_ * D_;       // (B,H,L,S)

    dim3 grid((B_ * H_ * LT_) / WAVES);   // 1024 blocks
    dim3 block(WAVES * 32);               // 4 wave32s, all same l-tile
    resattn_fwd_kernel<<<grid, block, 0, stream>>>(q, k, v, Ab, outV, outA);
}
